// forwardmodel_48352741818633
// MI455X (gfx1250) — matrix operand order repaired and checked by reading, not compiled, not run on hardware
//
#include <hip/hip_runtime.h>

// ============================================================================
// MI455X (gfx1250) implementation of the GIB-GCN drug-target model.
//
// Design (compile-only; reasoned from MI455X specs):
//  * All heavy GEMMs use v_wmma_f32_16x16x32_bf16 (wave32, 16x16 tiles,
//    f32 accumulate). Weights are pre-packed into a contiguous-k (16*half+e) B-fragment lane
//    layout so each fragment load is one contiguous 32B read.
//  * cl1 over [64,2000,512] is algebraically factored (emb is batch-invariant,
//    mol_con is node-invariant) -> one [2000,256]x[256,256] GEMM + one
//    [64,256]x[256,256] GEMM, avoiding 262MB of h materialization.
//  * The real 16.8 GFLOP GEMM (cl layer over 128000 rows) is fused:
//    h1 built in LDS (bf16), WMMA GEMM vs cl_W, cl2 (256->2) folded into the
//    epilogue, softmax + pos_emb accumulation in the same kernel.
//  * Everything else (scatter-GCN, BN, pooling, new_adj, heads) is tiny
//    (<0.3 GFLOP total) -> simple VALU kernels.
//
// Input order assumption: insertion-order flattening of setup_inputs():
//   0 protein_node_feat [2000,128]   1 protein_edge_index [2,32000]
//   2 node_feat [2048,9]             3 edge_index [2,4096]
//   4 batch [2048]
//   5..20  protein_gcn 4x(W,b,gamma,beta)
//   21..32 mol_gcn     3x(W,b,gamma,beta)
//   33     atom_emb [4609,128]
//   34..53 cl1,cl,cl2,gib_fc1,gib_fc2,fm_fc1,fm_fc2,mlp1,mlp2,mlp3 (W,b each)
// Output: d_out[0..63]=pred, [64]=pos_penalty, [65..576]=summary.
// ============================================================================

typedef __attribute__((ext_vector_type(16))) __bf16 v16bf;
typedef __attribute__((ext_vector_type(8)))  float  v8f;

#define NPn 2000
#define EPn 32000
#define NMn 2048
#define EMn 4096
#define CH  256
#define BGR 64

static __device__ __forceinline__ __bf16 f2bf(float f) {
  union { float f; unsigned u; } a; a.f = f;
  unsigned r = (a.u + 0x7FFFu + ((a.u >> 16) & 1u)) >> 16;  // RNE
  union { unsigned short s; __bf16 b; } o; o.s = (unsigned short)r;
  return o.b;
}

// ---------------------------------------------------------------------------
// Weight packing: f32 row-major W[K,N] -> bf16 B-fragments.
// Per 32x16 (KxN) tile: lane L (half=L/16, n=L%16) holds 16 contiguous bf16,
// element e -> K = kt*32 + half*16 + e  (contiguous k order, not the ISA table order).
// grid.x = (K/32)*(N/16), block = 32.
// ---------------------------------------------------------------------------
__global__ void pack_b_bf16(const float* __restrict__ W, int N,
                            __bf16* __restrict__ out) {
  int lane = threadIdx.x;
  int nnt = N >> 4;
  int kt = blockIdx.x / nnt, nt = blockIdx.x % nnt;
  int half = lane >> 4, lm = lane & 15;
  __bf16* dst = out + ((size_t)blockIdx.x * 32 + lane) * 16;
  int col = nt * 16 + lm;
#pragma unroll
  for (int e = 0; e < 16; ++e) {
    int kk = kt * 32 + half * 16 + e;
    dst[e] = f2bf(W[(size_t)kk * N + col]);
  }
}

// ---------------------------------------------------------------------------
// Generic WMMA GEMM: C[M,256] = A[M,K] @ Bpk.  K in {128,256} (template).
// Block = 256 threads (8 waves), 128-row tile; wave w owns rows 16w..16w+15.
// A tile staged in LDS as bf16; A-fragment gather matches the B packing:
// K(e,half) = half*16+e.
// ---------------------------------------------------------------------------
template <int K>
__global__ __launch_bounds__(256) void gemm_bf16(const float* __restrict__ A,
                                                 int M,
                                                 const __bf16* __restrict__ Bpk,
                                                 float* __restrict__ C) {
  __shared__ __bf16 As[128 * K];
  const int tid = threadIdx.x;
  const int row0 = blockIdx.x * 128;
  for (int i = tid; i < 128 * K; i += 256) {
    int m = i / K, k = i - m * K;
    int r = row0 + m;
    As[i] = f2bf(r < M ? A[(size_t)r * K + k] : 0.0f);
  }
  __syncthreads();
  const int wave = tid >> 5, lane = tid & 31;
  const int half = lane >> 4, lm = lane & 15;
  const int nkt = K / 32;
  v16bf afrag[8];
#pragma unroll
  for (int kt = 0; kt < nkt; ++kt) {
    const __bf16* base = As + (wave * 16 + lm) * K + kt * 32;
    v16bf a;
#pragma unroll
    for (int e = 0; e < 16; ++e) {
      int kk = half * 16 + e;
      a[e] = base[kk];
    }
    afrag[kt] = a;
  }
  for (int nt = 0; nt < 16; ++nt) {
    v8f acc = {};
#pragma unroll
    for (int kt = 0; kt < nkt; ++kt) {
      const v16bf b = *(const v16bf*)(Bpk + (((size_t)kt * 16 + nt) * 32 + lane) * 16);
      acc = __builtin_amdgcn_wmma_f32_16x16x32_bf16(false, afrag[kt], false, b,
                                                    (short)0, acc, false, false);
    }
    int col = nt * 16 + lm;
#pragma unroll
    for (int v = 0; v < 8; ++v) {
      int r = row0 + wave * 16 + half * 8 + v;   // C layout: M = v + 8*half
      if (r < M) C[(size_t)r * 256 + col] = acc[v];
    }
  }
}

// ---------------------------------------------------------------------------
// Fused assignment kernel (the dominant GEMM):
//  h1 = relu(prot_part[n] + mol_part[b] + cl1_b)      (LDS, bf16)
//  h2 = relu(h1 @ cl_W + cl_b)                        (WMMA, kept in regs)
//  logits = h2 @ cl2_W + cl2_b  -> softmax -> assignment[b,n,0..1]
//  pos_sum[b,c] += sum_n a0[n]*prot[n,c]
// grid = (16 node tiles, 64 graphs), block = 256.
// ---------------------------------------------------------------------------
__global__ __launch_bounds__(256) void assign_fused(
    const float* __restrict__ prot_part, const float* __restrict__ mol_part,
    const float* __restrict__ cl1_b, const __bf16* __restrict__ clW_pk,
    const float* __restrict__ cl_b, const float* __restrict__ cl2_W,
    const float* __restrict__ cl2_b, const float* __restrict__ prot,
    float* __restrict__ assignment, float* __restrict__ pos_sum) {
  __shared__ __bf16 h1s[128 * 256];
  __shared__ float logit[128 * 2];
  __shared__ float asgs[128 * 2];
  const int b = blockIdx.y, row0 = blockIdx.x * 128;
  const int tid = threadIdx.x;
  for (int i = tid; i < 128 * 256; i += 256) {
    int m = i >> 8, k = i & 255;
    int r = row0 + m;
    float v = 0.0f;
    if (r < NPn)
      v = fmaxf(prot_part[(size_t)r * 256 + k] + mol_part[b * 256 + k] + cl1_b[k], 0.0f);
    h1s[i] = f2bf(v);
  }
  if (tid < 256) logit[tid & 255] = 0.0f;
  __syncthreads();
  const int wave = tid >> 5, lane = tid & 31;
  const int half = lane >> 4, lm = lane & 15;
  v16bf afrag[8];
#pragma unroll
  for (int kt = 0; kt < 8; ++kt) {
    const __bf16* base = h1s + (wave * 16 + lm) * 256 + kt * 32;
    v16bf a;
#pragma unroll
    for (int e = 0; e < 16; ++e) {
      int kk = half * 16 + e;
      a[e] = base[kk];
    }
    afrag[kt] = a;
  }
  float l0[8], l1[8];
#pragma unroll
  for (int v = 0; v < 8; ++v) { l0[v] = 0.0f; l1[v] = 0.0f; }
  for (int nt = 0; nt < 16; ++nt) {
    v8f acc = {};
#pragma unroll
    for (int kt = 0; kt < 8; ++kt) {
      const v16bf bfr = *(const v16bf*)(clW_pk + (((size_t)kt * 16 + nt) * 32 + lane) * 16);
      acc = __builtin_amdgcn_wmma_f32_16x16x32_bf16(false, afrag[kt], false, bfr,
                                                    (short)0, acc, false, false);
    }
    int col = nt * 16 + lm;
    float cb = cl_b[col];
    float w0 = cl2_W[col * 2 + 0], w1 = cl2_W[col * 2 + 1];
#pragma unroll
    for (int v = 0; v < 8; ++v) {
      float h2 = fmaxf(acc[v] + cb, 0.0f);
      l0[v] += h2 * w0;
      l1[v] += h2 * w1;
    }
  }
#pragma unroll
  for (int v = 0; v < 8; ++v) {
    int mrow = wave * 16 + half * 8 + v;
    atomicAdd(&logit[mrow * 2 + 0], l0[v]);
    atomicAdd(&logit[mrow * 2 + 1], l1[v]);
  }
  __syncthreads();
  if (tid < 128) {
    float g0 = logit[tid * 2 + 0] + cl2_b[0];
    float g1 = logit[tid * 2 + 1] + cl2_b[1];
    float mx = fmaxf(g0, g1);
    float e0 = __expf(g0 - mx), e1 = __expf(g1 - mx);
    float inv = 1.0f / (e0 + e1);
    float a0 = e0 * inv, a1 = e1 * inv;
    asgs[tid * 2 + 0] = a0;
    asgs[tid * 2 + 1] = a1;
    int r = row0 + tid;
    if (r < NPn) {
      assignment[((size_t)b * NPn + r) * 2 + 0] = a0;
      assignment[((size_t)b * NPn + r) * 2 + 1] = a1;
    }
  }
  __syncthreads();
  int rmax = NPn - row0; rmax = rmax > 128 ? 128 : rmax;
  if (rmax > 0) {
    float s = 0.0f;
    for (int m = 0; m < rmax; ++m)
      s += asgs[m * 2] * prot[(size_t)(row0 + m) * 256 + tid];
    atomicAdd(&pos_sum[b * 256 + tid], s);
  }
}

// ------------------------------- small kernels ------------------------------
__global__ void fill_f32(float* p, float v, int n) {
  int i = blockIdx.x * 256 + threadIdx.x;
  if (i < n) p[i] = v;
}
__global__ void scale_ip(float* p, float s, int n) {
  int i = blockIdx.x * 256 + threadIdx.x;
  if (i < n) p[i] *= s;
}
__global__ void count_deg(const int* __restrict__ dst, int E, float* deg) {
  int e = blockIdx.x * 256 + threadIdx.x;
  if (e < E) atomicAdd(&deg[dst[e]], 1.0f);
}
__global__ void rsqrt_ip(float* p, int n) {
  int i = blockIdx.x * 256 + threadIdx.x;
  if (i < n) p[i] = rsqrtf(p[i]);
}
__global__ void atom_encode(const int* __restrict__ nfeat,
                            const float* __restrict__ emb,
                            float* __restrict__ out) {
  int n = blockIdx.x, d = threadIdx.x;  // block = 128
  float s = 0.0f;
#pragma unroll
  for (int j = 0; j < 9; ++j) s += emb[(size_t)nfeat[n * 9 + j] * 128 + d];
  out[(size_t)n * 128 + d] = s * (1.0f / 9.0f);
}
// out[n,c] = h[n,c]*dis[n]^2 + bias[c]   (self-loop term + conv bias)
__global__ void gcn_self_bias(const float* __restrict__ h,
                              const float* __restrict__ dis,
                              const float* __restrict__ bias,
                              float* __restrict__ out, int M) {
  int i = blockIdx.x * 256 + threadIdx.x;
  if (i >= M * 256) return;
  int r = i >> 8, c = i & 255;
  float d = dis[r];
  out[i] = h[i] * d * d + bias[c];
}
// out[dst] += h[src]*dis[src]*dis[dst]   (one block per edge, 256 channels)
__global__ void gcn_scatter(const float* __restrict__ h,
                            const int* __restrict__ src,
                            const int* __restrict__ dst,
                            const float* __restrict__ dis,
                            float* __restrict__ out) {
  int e = blockIdx.x, c = threadIdx.x;
  int s = src[e], d = dst[e];
  float nrm = dis[s] * dis[d];
  atomicAdd(&out[(size_t)d * 256 + c], h[(size_t)s * 256 + c] * nrm);
}
__global__ void bn_stats(const float* __restrict__ X, int M,
                         float* __restrict__ mv) {
  __shared__ float s1[256], s2[256];
  int c = blockIdx.x;
  float sum = 0.0f, sq = 0.0f;
  for (int r = threadIdx.x; r < M; r += 256) {
    float v = X[(size_t)r * 256 + c];
    sum += v; sq += v * v;
  }
  s1[threadIdx.x] = sum; s2[threadIdx.x] = sq;
  __syncthreads();
  for (int o = 128; o > 0; o >>= 1) {
    if (threadIdx.x < o) {
      s1[threadIdx.x] += s1[threadIdx.x + o];
      s2[threadIdx.x] += s2[threadIdx.x + o];
    }
    __syncthreads();
  }
  if (threadIdx.x == 0) {
    float m = s1[0] / (float)M;
    float var = s2[0] / (float)M - m * m;
    mv[c] = m;
    mv[256 + c] = rsqrtf(var + 1e-5f);
  }
}
__global__ void bn_apply(float* __restrict__ X, int M,
                         const float* __restrict__ mv,
                         const float* __restrict__ g,
                         const float* __restrict__ bt, int relu) {
  int i = blockIdx.x * 256 + threadIdx.x;
  if (i >= M * 256) return;
  int c = i & 255;
  float v = (X[i] - mv[c]) * mv[256 + c] * g[c] + bt[c];
  X[i] = relu ? fmaxf(v, 0.0f) : v;
}
__global__ void pool_acc(const float* __restrict__ mol,
                         const int* __restrict__ batch,
                         float* __restrict__ sum, float* __restrict__ cnt) {
  int n = blockIdx.x, c = threadIdx.x;
  int g = batch[n];
  atomicAdd(&sum[(size_t)g * 256 + c], mol[(size_t)n * 256 + c]);
  if (c == 0) atomicAdd(&cnt[g], 1.0f);
}
__global__ void pool_div(float* __restrict__ sum, const float* __restrict__ cnt) {
  int i = blockIdx.x * 256 + threadIdx.x;
  if (i >= BGR * 256) return;
  sum[i] /= fmaxf(cnt[i >> 8], 1.0f);
}
// Per-row dense layer for [64, *] matrices.
__global__ __launch_bounds__(256) void linear_small(
    const float* __restrict__ A, int lda, const float* __restrict__ W, int ldw,
    const float* __restrict__ bias, float* __restrict__ C, int ldc, int K,
    int N, int relu) {
  int row = blockIdx.x;
  const float* a = A + (size_t)row * lda;
  for (int n = threadIdx.x; n < N; n += 256) {
    float acc = bias ? bias[n] : 0.0f;
    for (int k = 0; k < K; ++k) acc += a[k] * W[(size_t)k * ldw + n];
    if (relu) acc = fmaxf(acc, 0.0f);
    C[(size_t)row * ldc + n] = acc;
  }
}
// new_adj[b,c,d] = sum_edges a[b,src,c]*a[b,dst,d]
__global__ void new_adj_kernel(const float* __restrict__ asg,
                               const int* __restrict__ src,
                               const int* __restrict__ dst,
                               float* __restrict__ nadj) {
  __shared__ float r0[256], r1[256], r2[256], r3[256];
  int b = blockIdx.x;
  float a00 = 0, a01 = 0, a10 = 0, a11 = 0;
  for (int e = threadIdx.x; e < EPn; e += 256) {
    int s = src[e], d = dst[e];
    float s0 = asg[((size_t)b * NPn + s) * 2 + 0];
    float s1 = asg[((size_t)b * NPn + s) * 2 + 1];
    float d0 = asg[((size_t)b * NPn + d) * 2 + 0];
    float d1 = asg[((size_t)b * NPn + d) * 2 + 1];
    a00 += s0 * d0; a01 += s0 * d1; a10 += s1 * d0; a11 += s1 * d1;
  }
  r0[threadIdx.x] = a00; r1[threadIdx.x] = a01;
  r2[threadIdx.x] = a10; r3[threadIdx.x] = a11;
  __syncthreads();
  for (int o = 128; o > 0; o >>= 1) {
    if (threadIdx.x < o) {
      r0[threadIdx.x] += r0[threadIdx.x + o];
      r1[threadIdx.x] += r1[threadIdx.x + o];
      r2[threadIdx.x] += r2[threadIdx.x + o];
      r3[threadIdx.x] += r3[threadIdx.x + o];
    }
    __syncthreads();
  }
  if (threadIdx.x == 0) {
    nadj[b * 4 + 0] = r0[0]; nadj[b * 4 + 1] = r1[0];
    nadj[b * 4 + 2] = r2[0]; nadj[b * 4 + 3] = r3[0];
  }
}
__global__ void penalty_kernel(const float* __restrict__ nadj,
                               float* __restrict__ out) {
  __shared__ float red[64];
  int b = threadIdx.x;  // block = 64
  float n00 = nadj[b * 4 + 0], n01 = nadj[b * 4 + 1];
  float n10 = nadj[b * 4 + 2], n11 = nadj[b * 4 + 3];
  float d0 = n00 / fmaxf(fabsf(n00) + fabsf(n01), 1e-5f);
  float d1 = n11 / fmaxf(fabsf(n10) + fabsf(n11), 1e-5f);
  red[b] = (d0 - 1.0f) * (d0 - 1.0f) + (d1 - 1.0f) * (d1 - 1.0f);
  __syncthreads();
  for (int o = 32; o > 0; o >>= 1) {
    if (b < o) red[b] += red[b + o];
    __syncthreads();
  }
  if (b == 0) out[0] = red[0] / 128.0f;  // mean over [B,2]
}
__global__ void summary_kernel(const float* __restrict__ pos_emb,
                               const float* __restrict__ mol_emb,
                               float* __restrict__ out) {
  int c = blockIdx.x * 256 + threadIdx.x;
  if (c >= 512) return;
  const float* p = (c < 256) ? (pos_emb + c) : (mol_emb + (c - 256));
  float s = 0.0f;
  for (int b = 0; b < BGR; ++b) s += p[(size_t)b * 256];
  out[c] = s / (float)BGR;
}

// ============================================================================
extern "C" void kernel_launch(void* const* d_in, const int* in_sizes, int n_in,
                              void* d_out, int out_size, void* d_ws,
                              size_t ws_size, hipStream_t stream) {
  (void)in_sizes; (void)n_in; (void)out_size; (void)ws_size;
  const float* prot_feat = (const float*)d_in[0];
  const int* pei = (const int*)d_in[1];
  const int* nfeat = (const int*)d_in[2];
  const int* mei = (const int*)d_in[3];
  const int* batch = (const int*)d_in[4];
  int idx = 5;
  const float *pW[4], *pB[4], *pG[4], *pBt[4];
  for (int l = 0; l < 4; ++l) {
    pW[l] = (const float*)d_in[idx++]; pB[l] = (const float*)d_in[idx++];
    pG[l] = (const float*)d_in[idx++]; pBt[l] = (const float*)d_in[idx++];
  }
  const float *mW[3], *mB[3], *mG[3], *mBt[3];
  for (int l = 0; l < 3; ++l) {
    mW[l] = (const float*)d_in[idx++]; mB[l] = (const float*)d_in[idx++];
    mG[l] = (const float*)d_in[idx++]; mBt[l] = (const float*)d_in[idx++];
  }
  const float* atom_emb = (const float*)d_in[idx++];
  const float* cl1_W = (const float*)d_in[idx++];
  const float* cl1_b = (const float*)d_in[idx++];
  const float* cl_W  = (const float*)d_in[idx++];
  const float* cl_b  = (const float*)d_in[idx++];
  const float* cl2_W = (const float*)d_in[idx++];
  const float* cl2_b = (const float*)d_in[idx++];
  const float* gib1_W = (const float*)d_in[idx++];
  const float* gib1_b = (const float*)d_in[idx++];
  const float* gib2_W = (const float*)d_in[idx++];
  const float* gib2_b = (const float*)d_in[idx++];
  const float* fm1_W = (const float*)d_in[idx++];
  const float* fm1_b = (const float*)d_in[idx++];
  const float* fm2_W = (const float*)d_in[idx++];
  const float* fm2_b = (const float*)d_in[idx++];
  const float* mlp1_W = (const float*)d_in[idx++];
  const float* mlp1_b = (const float*)d_in[idx++];
  const float* mlp2_W = (const float*)d_in[idx++];
  const float* mlp2_b = (const float*)d_in[idx++];
  const float* mlp3_W = (const float*)d_in[idx++];
  const float* mlp3_b = (const float*)d_in[idx++];
  float* out = (float*)d_out;

  char* base = (char*)d_ws;
  size_t off = 0;
  auto alloc = [&](size_t bytes) -> void* {
    off = (off + 255) & ~(size_t)255;
    void* p = base + off;
    off += bytes;
    return p;
  };

  // ---- pack weights into WMMA B-fragment layout (bf16) ----
  __bf16* pk_p[4]; __bf16* pk_m[3];
  pk_p[0] = (__bf16*)alloc(128 * 256 * 2);
  pack_b_bf16<<<(128 / 32) * 16, 32, 0, stream>>>(pW[0], 256, pk_p[0]);
  for (int l = 1; l < 4; ++l) {
    pk_p[l] = (__bf16*)alloc(256 * 256 * 2);
    pack_b_bf16<<<(256 / 32) * 16, 32, 0, stream>>>(pW[l], 256, pk_p[l]);
  }
  pk_m[0] = (__bf16*)alloc(128 * 256 * 2);
  pack_b_bf16<<<(128 / 32) * 16, 32, 0, stream>>>(mW[0], 256, pk_m[0]);
  for (int l = 1; l < 3; ++l) {
    pk_m[l] = (__bf16*)alloc(256 * 256 * 2);
    pack_b_bf16<<<(256 / 32) * 16, 32, 0, stream>>>(mW[l], 256, pk_m[l]);
  }
  __bf16* pk_cl1top = (__bf16*)alloc(256 * 256 * 2);  // rows 0..255 of cl1_W
  pack_b_bf16<<<(256 / 32) * 16, 32, 0, stream>>>(cl1_W, 256, pk_cl1top);
  __bf16* pk_clW = (__bf16*)alloc(256 * 256 * 2);
  pack_b_bf16<<<(256 / 32) * 16, 32, 0, stream>>>(cl_W, 256, pk_clW);

  // ---- degrees ----
  float* pdis = (float*)alloc(NPn * 4);
  fill_f32<<<(NPn + 255) / 256, 256, 0, stream>>>(pdis, 1.0f, NPn);
  count_deg<<<(EPn + 255) / 256, 256, 0, stream>>>(pei + EPn, EPn, pdis);
  rsqrt_ip<<<(NPn + 255) / 256, 256, 0, stream>>>(pdis, NPn);
  float* mdis = (float*)alloc(NMn * 4);
  fill_f32<<<(NMn + 255) / 256, 256, 0, stream>>>(mdis, 1.0f, NMn);
  count_deg<<<(EMn + 255) / 256, 256, 0, stream>>>(mei + EMn, EMn, mdis);
  rsqrt_ip<<<(NMn + 255) / 256, 256, 0, stream>>>(mdis, NMn);

  // ---- buffers ----
  float* nf = (float*)alloc((size_t)NMn * 128 * 4);
  float* pbA = (float*)alloc((size_t)NPn * 256 * 4);
  float* pbB = (float*)alloc((size_t)NPn * 256 * 4);
  float* gtmp = (float*)alloc((size_t)NMn * 256 * 4);  // big enough for both
  float* mbA = (float*)alloc((size_t)NMn * 256 * 4);
  float* mbB = (float*)alloc((size_t)NMn * 256 * 4);
  float* mv = (float*)alloc(2 * 256 * 4);
  float* mol_emb = (float*)alloc(BGR * 256 * 4);
  float* cnt = (float*)alloc(BGR * 4);
  float* prot_part = (float*)alloc((size_t)NPn * 256 * 4);
  float* mol_part = (float*)alloc(BGR * 256 * 4);
  float* assignment = (float*)alloc((size_t)BGR * NPn * 2 * 4);
  float* pos_emb = (float*)alloc(BGR * 256 * 4);
  float* nadj = (float*)alloc(BGR * 4 * 4);
  float* t1 = (float*)alloc(BGR * 256 * 4);
  float* zc = (float*)alloc(BGR * 512 * 4);
  float* z1 = (float*)alloc(BGR * 512 * 4);
  float* z2 = (float*)alloc(BGR * 512 * 4);

  // ---- GCN layer driver ----
  auto gcn_layer = [&](const float* x, int M, int K, const __bf16* wpk,
                       const float* bias, const float* g, const float* bt,
                       const int* src, const int* dst, const float* dis, int E,
                       float* obuf, int relu) {
    if (K == 128)
      gemm_bf16<128><<<(M + 127) / 128, 256, 0, stream>>>(x, M, wpk, gtmp);
    else
      gemm_bf16<256><<<(M + 127) / 128, 256, 0, stream>>>(x, M, wpk, gtmp);
    int total = M * 256;
    gcn_self_bias<<<(total + 255) / 256, 256, 0, stream>>>(gtmp, dis, bias, obuf, M);
    gcn_scatter<<<E, 256, 0, stream>>>(gtmp, src, dst, dis, obuf);
    bn_stats<<<256, 256, 0, stream>>>(obuf, M, mv);
    bn_apply<<<(total + 255) / 256, 256, 0, stream>>>(obuf, M, mv, g, bt, relu);
  };

  // ---- protein GCN (4 layers; last without relu) ----
  gcn_layer(prot_feat, NPn, 128, pk_p[0], pB[0], pG[0], pBt[0], pei, pei + EPn, pdis, EPn, pbA, 1);
  gcn_layer(pbA, NPn, 256, pk_p[1], pB[1], pG[1], pBt[1], pei, pei + EPn, pdis, EPn, pbB, 1);
  gcn_layer(pbB, NPn, 256, pk_p[2], pB[2], pG[2], pBt[2], pei, pei + EPn, pdis, EPn, pbA, 1);
  gcn_layer(pbA, NPn, 256, pk_p[3], pB[3], pG[3], pBt[3], pei, pei + EPn, pdis, EPn, pbB, 0);
  float* protX = pbB;  // [2000,256]

  // ---- molecule: atom encoder + GCN (3 layers; last without relu) ----
  atom_encode<<<NMn, 128, 0, stream>>>(nfeat, atom_emb, nf);
  gcn_layer(nf, NMn, 128, pk_m[0], mB[0], mG[0], mBt[0], mei, mei + EMn, mdis, EMn, mbA, 1);
  gcn_layer(mbA, NMn, 256, pk_m[1], mB[1], mG[1], mBt[1], mei, mei + EMn, mdis, EMn, mbB, 1);
  gcn_layer(mbB, NMn, 256, pk_m[2], mB[2], mG[2], mBt[2], mei, mei + EMn, mdis, EMn, mbA, 0);
  float* molX = mbA;  // [2048,256]

  // ---- global mean pool ----
  fill_f32<<<(BGR * 256 + 255) / 256, 256, 0, stream>>>(mol_emb, 0.0f, BGR * 256);
  fill_f32<<<1, 256, 0, stream>>>(cnt, 0.0f, BGR);
  pool_acc<<<NMn, 256, 0, stream>>>(molX, batch, mol_emb, cnt);
  pool_div<<<(BGR * 256 + 255) / 256, 256, 0, stream>>>(mol_emb, cnt);

  // ---- cl1 factorization ----
  gemm_bf16<256><<<(NPn + 127) / 128, 256, 0, stream>>>(protX, NPn, pk_cl1top, prot_part);
  linear_small<<<BGR, 256, 0, stream>>>(mol_emb, 256, cl1_W + 256 * 256, 256,
                                        nullptr, mol_part, 256, 256, 256, 0);

  // ---- fused assignment (dominant WMMA GEMM) ----
  fill_f32<<<(BGR * 256 + 255) / 256, 256, 0, stream>>>(pos_emb, 0.0f, BGR * 256);
  assign_fused<<<dim3(16, BGR), 256, 0, stream>>>(prot_part, mol_part, cl1_b,
                                                  pk_clW, cl_b, cl2_W, cl2_b,
                                                  protX, assignment, pos_emb);
  scale_ip<<<(BGR * 256 + 255) / 256, 256, 0, stream>>>(pos_emb, 1.0f / (float)NPn, BGR * 256);

  // ---- new_adj + penalty ----
  new_adj_kernel<<<BGR, 256, 0, stream>>>(assignment, pei, pei + EPn, nadj);
  penalty_kernel<<<1, 64, 0, stream>>>(nadj, out + 64);

  // ---- dense heads ----
  linear_small<<<BGR, 256, 0, stream>>>(pos_emb, 256, gib1_W, 256, gib1_b, t1, 256, 256, 256, 1);
  linear_small<<<BGR, 256, 0, stream>>>(t1, 256, gib2_W, 256, gib2_b, zc, 512, 256, 256, 0);
  linear_small<<<BGR, 256, 0, stream>>>(mol_emb, 256, fm1_W, 256, fm1_b, t1, 256, 256, 256, 1);
  linear_small<<<BGR, 256, 0, stream>>>(t1, 256, fm2_W, 256, fm2_b, zc + 256, 512, 256, 256, 0);
  linear_small<<<BGR, 256, 0, stream>>>(zc, 512, mlp1_W, 512, mlp1_b, z1, 512, 512, 512, 1);
  linear_small<<<BGR, 256, 0, stream>>>(z1, 512, mlp2_W, 512, mlp2_b, z2, 512, 512, 512, 1);
  linear_small<<<BGR, 256, 0, stream>>>(z2, 512, mlp3_W, 1, mlp3_b, out, 1, 512, 1, 0);

  // ---- summary ----
  summary_kernel<<<2, 256, 0, stream>>>(pos_emb, mol_emb, out + 65);
}
